// SelfAttention_79525614452805
// MI455X (gfx1250) — matrix operand order repaired and checked by reading, not compiled, not run on hardware
//
#include <hip/hip_runtime.h>
#include <hip/hip_bf16.h>

// ---------------------------------------------------------------------------
// Self-attention forward for MI455X (gfx1250), bf16 WMMA path.
//   B=2, S=2048, D=1024, H=16, Dh=64
// Pipeline:
//   1) cast x -> bf16 row-major [M=4096, K=1024]
//   2) pack W{q,k,v,o} (fp32 [N][K], y = x @ W^T) into B-fragment-swizzled bf16
//   3) GEMM (WMMA bf16, f32 acc):
//        q: scaled by 1/sqrt(Dh) at write (folds softmax scale)
//        k: row-major bf16
//        v: written directly in attention B-fragment-packed layout
//   4) flash-attention per 16-row query tile (WMMA for QK^T and PV,
//      online softmax, P transposed via permuted LDS -> ds_load_b128)
//   5) GEMM with Wo (+bias) -> fp32 d_out
// ---------------------------------------------------------------------------

#define BATCH   2
#define SEQ     2048
#define DMODEL  1024
#define NHEADS  16
#define DHEAD   64
#define MROWS   (BATCH * SEQ)      // 4096

typedef __bf16 v16bf __attribute__((ext_vector_type(16)));
typedef float  v8f   __attribute__((ext_vector_type(8)));

union Frag {
    v16bf    v;
    __bf16   e[16];
    unsigned u[8];
    uint4    q[2];
};

// A-matrix (16-bit, 16x32) per-lane column index for dword dw (holds K=kc,kc+1)
__device__ __forceinline__ int a_kcol(int dw, int g) {
    int e = 2 * dw;
    return e + 16 * g;
}

// ---------------------------------------------------------------------------
// Cast fp32 -> bf16 (x activation)
__global__ void cast_f32_bf16(const float* __restrict__ in,
                              __bf16* __restrict__ out, int n) {
    int i = blockIdx.x * blockDim.x + threadIdx.x;
    if (i < n) out[i] = (__bf16)in[i];
}

// ---------------------------------------------------------------------------
// Pack W (fp32, [N][K], used as y = x @ W^T) into bf16 B-fragment layout:
// tile (kt, nt): 32 K-values x 16 N-cols; element (lane l, e):
//   value = W[nt*16 + (l&15)][kt*32 + e + 16*(l>>4)]
__global__ void pack_weight(const float* __restrict__ W,
                            __bf16* __restrict__ out, int N, int K) {
    int idx = blockIdx.x * blockDim.x + threadIdx.x;
    if (idx >= N * K) return;
    int e    = idx & 15;
    int l    = (idx >> 4) & 31;
    int tile = idx >> 9;
    int ntn  = N / 16;
    int nt   = tile % ntn;
    int kt   = tile / ntn;
    int kk   = kt * 32 + e + 16 * (l >> 4);
    int nn   = nt * 16 + (l & 15);
    out[idx] = (__bf16)W[(size_t)nn * K + kk];
}

// ---------------------------------------------------------------------------
// GEMM: out[M,N] = (A[M,K](bf16) * Bp(packed bf16) + bias) * oscale
// One wave per 32x64 output tile (2 A-frags, 8 accumulators).
// OUT_MODE: 0 -> bf16 row-major, 1 -> fp32 row-major,
//           2 -> bf16 packed into attention V B-fragment layout
//                [b][h][S/32][4][lane*16+e]
template <int OUT_MODE>
__global__ __launch_bounds__(32) void gemm_bf16(
    const __bf16* __restrict__ A, const __bf16* __restrict__ Bp,
    const float* __restrict__ bias, void* __restrict__ out,
    int M, int N, int K, float oscale) {
    const int lane = threadIdx.x & 31;
    const int g    = lane >> 4;
    const int ln   = lane & 15;
    const int nb   = blockIdx.x * 64;   // N tile base
    const int mb   = blockIdx.y * 32;   // M tile base (2 x 16)

    v8f acc[2][4];
#pragma unroll
    for (int mh = 0; mh < 2; ++mh)
#pragma unroll
        for (int n = 0; n < 4; ++n) acc[mh][n] = (v8f){0, 0, 0, 0, 0, 0, 0, 0};

    const __bf16* arow0  = A + (size_t)(mb + ln) * K;
    const __bf16* arow1  = A + (size_t)(mb + 16 + ln) * K;
    const int     ntiles = N / 16;

    for (int kt = 0; kt < K / 32; ++kt) {
        Frag fa0, fa1;
#pragma unroll
        for (int dw = 0; dw < 8; ++dw) {
            int kc     = kt * 32 + a_kcol(dw, g);
            fa0.u[dw]  = *(const unsigned*)(arow0 + kc);
            fa1.u[dw]  = *(const unsigned*)(arow1 + kc);
        }
        __builtin_prefetch(arow0 + (kt + 1) * 32, 0, 0);
        __builtin_prefetch(arow1 + (kt + 1) * 32, 0, 0);
#pragma unroll
        for (int n = 0; n < 4; ++n) {
            const __bf16* bp =
                Bp + ((size_t)(kt * ntiles + (nb >> 4) + n)) * 512 + lane * 16;
            Frag fb;
            fb.q[0]    = *(const uint4*)(bp);
            fb.q[1]    = *(const uint4*)(bp + 8);
            acc[0][n]  = __builtin_amdgcn_wmma_f32_16x16x32_bf16(
                false, fa0.v, false, fb.v, (short)0, acc[0][n], false, false);
            acc[1][n]  = __builtin_amdgcn_wmma_f32_16x16x32_bf16(
                false, fa1.v, false, fb.v, (short)0, acc[1][n], false, false);
        }
    }

#pragma unroll
    for (int mh = 0; mh < 2; ++mh)
#pragma unroll
        for (int n = 0; n < 4; ++n)
#pragma unroll
            for (int r = 0; r < 8; ++r) {
                int   row = mb + mh * 16 + r + 8 * g;
                int   col = nb + n * 16 + ln;
                float val = (acc[mh][n][r] + bias[col]) * oscale;
                if (OUT_MODE == 1) {
                    ((float*)out)[(size_t)row * N + col] = val;
                } else if (OUT_MODE == 0) {
                    ((__bf16*)out)[(size_t)row * N + col] = (__bf16)val;
                } else {
                    // scatter into attention V B-fragment layout
                    int    b   = row >> 11;          // / SEQ
                    int    s   = row & (SEQ - 1);
                    int    h   = col >> 6;           // / DHEAD
                    int    dh  = col & (DHEAD - 1);
                    int    tt  = s >> 5;
                    int    r32 = s & 31;
                    int    lp  = ((r32 >> 4) << 4) | (dh & 15);
                    int    ep  = r32 & 15;
                    int    nd  = dh >> 4;
                    size_t adr =
                        ((((size_t)(b * NHEADS + h) * (SEQ / 32) + tt) * 4 +
                          nd) << 9) + lp * 16 + ep;
                    ((__bf16*)out)[adr] = (__bf16)val;
                }
            }
}

// ---------------------------------------------------------------------------
// Flash attention: one wave handles one (b, h, 16-row query tile).
// q is pre-scaled by 1/sqrt(Dh); v is pre-packed in B-fragment layout.
__global__ __launch_bounds__(128) void flash_attn(
    const __bf16* __restrict__ q, const __bf16* __restrict__ k,
    const __bf16* __restrict__ vpk, __bf16* __restrict__ o) {
    __shared__ __bf16 plds[4][16 * 32];

    const int lane = threadIdx.x & 31;
    const int wave = threadIdx.x >> 5;
    const int g    = lane >> 4;
    const int ln   = lane & 15;

    int       tile  = blockIdx.x * 4 + wave;
    const int qtph  = SEQ / 16;           // query tiles per head
    int       head  = tile / qtph;        // 0 .. B*H-1
    int       qt    = tile % qtph;
    int       b     = head / NHEADS;
    int       h     = head % NHEADS;
    int       srow0 = qt * 16;

    const __bf16* qb = q + (size_t)(b * SEQ) * DMODEL + h * DHEAD;
    const __bf16* kb = k + (size_t)(b * SEQ) * DMODEL + h * DHEAD;
    const __bf16* vh = vpk + ((size_t)(b * NHEADS + h) * (SEQ / 32) * 4) * 512;

    // Q A-fragments for the two 32-wide d-chunks of Dh=64 (loaded once)
    Frag          qa[2];
    const __bf16* qrow = qb + (size_t)(srow0 + ln) * DMODEL;
#pragma unroll
    for (int c = 0; c < 2; ++c)
#pragma unroll
        for (int dw = 0; dw < 8; ++dw)
            qa[c].u[dw] = *(const unsigned*)(qrow + c * 32 + a_kcol(dw, g));

    float mrow[8], lrow[8];
    v8f   acc[4];
#pragma unroll
    for (int r = 0; r < 8; ++r) { mrow[r] = -1e30f; lrow[r] = 0.f; }
#pragma unroll
    for (int n = 0; n < 4; ++n) acc[n] = (v8f){0, 0, 0, 0, 0, 0, 0, 0};

    // LDS column for P writes: p(t) = t (contiguous k order)
    const int p0c = ln;                              // t = ln
    const int p1c = 16 + ln;                         // t = 16 + ln

    for (int j = 0; j < SEQ / 32; ++j) {
        int tbase = j * 32;

        // ---- scores: S = Q * K^T (Q already carries 1/sqrt(Dh)) ----
        v8f sc[2];
#pragma unroll
        for (int n = 0; n < 2; ++n) {
            sc[n]              = (v8f){0, 0, 0, 0, 0, 0, 0, 0};
            int           trow = tbase + n * 16 + ln;  // key row (B column)
            const __bf16* krow = kb + (size_t)trow * DMODEL;
#pragma unroll
            for (int c = 0; c < 2; ++c) {
                Frag          fb;
                const __bf16* p = krow + c * 32 + 16 * g;  // 16 contiguous d
                fb.q[0]         = *(const uint4*)(p);
                fb.q[1]         = *(const uint4*)(p + 8);
                sc[n] = __builtin_amdgcn_wmma_f32_16x16x32_bf16(
                    false, qa[c].v, false, fb.v, (short)0, sc[n], false, false);
            }
        }

        // ---- online softmax update (row stats per VGPR r, row = r+8g) ----
        float mnew[8], alpha[8];
#pragma unroll
        for (int r = 0; r < 8; ++r) {
            float mx = fmaxf(sc[0][r], sc[1][r]);
#pragma unroll
            for (int d = 1; d < 16; d <<= 1)
                mx = fmaxf(mx, __shfl_xor(mx, d, 32));
            mnew[r]  = fmaxf(mrow[r], mx);
            alpha[r] = __expf(mrow[r] - mnew[r]);
            mrow[r]  = mnew[r];
        }
        float p0[8], p1[8];
#pragma unroll
        for (int r = 0; r < 8; ++r) {
            p0[r]      = __expf(sc[0][r] - mnew[r]);
            p1[r]      = __expf(sc[1][r] - mnew[r]);
            float srow = p0[r] + p1[r];
#pragma unroll
            for (int d = 1; d < 16; d <<= 1) srow += __shfl_xor(srow, d, 32);
            lrow[r] = lrow[r] * alpha[r] + srow;
#pragma unroll
            for (int n = 0; n < 4; ++n) acc[n][r] *= alpha[r];
        }

        // ---- P (C layout) -> permuted LDS so A-frag reads are contiguous ----
#pragma unroll
        for (int r = 0; r < 8; ++r) {
            int row = r + 8 * g;
            plds[wave][row * 32 + p0c] = (__bf16)p0[r];
            plds[wave][row * 32 + p1c] = (__bf16)p1[r];
        }
        __syncthreads();
        Frag pa;
        {
            const __bf16* pp = &plds[wave][ln * 32 + g * 16];
            pa.q[0] = *(const uint4*)(pp);
            pa.q[1] = *(const uint4*)(pp + 8);
        }
        __syncthreads();

        // ---- out += P * V (V pre-packed: 2 x b128 per fragment) ----
#pragma unroll
        for (int n = 0; n < 4; ++n) {
            const __bf16* vp = vh + (((size_t)j * 4 + n) << 9) + lane * 16;
            Frag          fv;
            fv.q[0] = *(const uint4*)(vp);
            fv.q[1] = *(const uint4*)(vp + 8);
            acc[n]  = __builtin_amdgcn_wmma_f32_16x16x32_bf16(
                false, pa.v, false, fv.v, (short)0, acc[n], false, false);
        }
    }

    // ---- epilogue: normalize and store bf16 ----
#pragma unroll
    for (int n = 0; n < 4; ++n)
#pragma unroll
        for (int r = 0; r < 8; ++r) {
            int row = srow0 + r + 8 * g;
            int col = h * DHEAD + n * 16 + ln;
            o[(size_t)(b * SEQ + row) * DMODEL + col] =
                (__bf16)(acc[n][r] / lrow[r]);
        }
}

// ---------------------------------------------------------------------------
extern "C" void kernel_launch(void* const* d_in, const int* in_sizes, int n_in,
                              void* d_out, int out_size, void* d_ws,
                              size_t ws_size, hipStream_t stream) {
    const float* x  = (const float*)d_in[0];
    const float* Wq = (const float*)d_in[1];
    const float* bq = (const float*)d_in[2];
    const float* Wk = (const float*)d_in[3];
    const float* bk = (const float*)d_in[4];
    const float* Wv = (const float*)d_in[5];
    const float* bv = (const float*)d_in[6];
    const float* Wo = (const float*)d_in[7];
    const float* bo = (const float*)d_in[8];

    char* ws = (char*)d_ws;
    const size_t MiB = 1ull << 20;
    __bf16* xb  = (__bf16*)(ws + 0 * MiB);    // 8 MiB  (4096x1024 bf16)
    __bf16* wqp = (__bf16*)(ws + 8 * MiB);    // 2 MiB each packed weight
    __bf16* wkp = (__bf16*)(ws + 10 * MiB);
    __bf16* wvp = (__bf16*)(ws + 12 * MiB);
    __bf16* wop = (__bf16*)(ws + 14 * MiB);
    __bf16* qb  = (__bf16*)(ws + 16 * MiB);   // 8 MiB (pre-scaled q)
    __bf16* kb  = (__bf16*)(ws + 24 * MiB);   // 8 MiB
    __bf16* vpk = (__bf16*)(ws + 32 * MiB);   // 8 MiB (B-fragment packed v)
    __bf16* ob  = (__bf16*)(ws + 40 * MiB);   // 8 MiB (attn output, pre-proj)

    const int D = DMODEL, M = MROWS;

    // 1) pack weights + cast activations
    {
        int  tot = D * D;
        dim3 blk(256), grd((tot + 255) / 256);
        pack_weight<<<grd, blk, 0, stream>>>(Wq, wqp, D, D);
        pack_weight<<<grd, blk, 0, stream>>>(Wk, wkp, D, D);
        pack_weight<<<grd, blk, 0, stream>>>(Wv, wvp, D, D);
        pack_weight<<<grd, blk, 0, stream>>>(Wo, wop, D, D);
        int  n = M * D;
        cast_f32_bf16<<<(n + 255) / 256, 256, 0, stream>>>(x, xb, n);
    }

    // 2) QKV projections (q carries the 1/sqrt(Dh) softmax scale)
    dim3 ggrid(D / 64, M / 32), gblk(32);
    gemm_bf16<0><<<ggrid, gblk, 0, stream>>>(xb, wqp, bq, qb, M, D, D, 0.125f);
    gemm_bf16<0><<<ggrid, gblk, 0, stream>>>(xb, wkp, bk, kb, M, D, D, 1.0f);
    gemm_bf16<2><<<ggrid, gblk, 0, stream>>>(xb, wvp, bv, vpk, M, D, D, 1.0f);

    // 3) flash attention: B*H*(S/16) = 4096 wave-tiles, 4 waves per block
    flash_attn<<<(BATCH * NHEADS * (SEQ / 16)) / 4, 128, 0, stream>>>(qb, kb,
                                                                      vpk, ob);

    // 4) output projection (fp32 out + bias)
    gemm_bf16<1><<<ggrid, gblk, 0, stream>>>(ob, wop, bo, d_out, M, D, D, 1.0f);
}
